// Classifier_36618891166176
// MI455X (gfx1250) — matrix operand order repaired and checked by reading, not compiled, not run on hardware
//
#include <hip/hip_runtime.h>
#include <hip/hip_bf16.h>

typedef __attribute__((ext_vector_type(16))) _Float16 v16h;
typedef __attribute__((ext_vector_type(8)))  _Float16 v8h;
typedef __attribute__((ext_vector_type(8)))  float    v8f;

namespace {
constexpr int NN = 100000;   // nodes
constexpr int EE = 1000000;  // edges per relation
constexpr int RR = 4;        // relations
constexpr int GG = 256;      // graphs
constexpr int DH = 128;      // hidden / in feats
constexpr int NC = 10;       // classes
constexpr int LL = 3;        // layers

constexpr int MODE_FIRST = 1;  // first relation: store (acc not yet valid)
constexpr int MODE_RELU  = 2;  // last relation: apply ReLU to the sum
}

// ---------------------------------------------------------------------------
// Degree counting: one thread per (relation, edge)
// ---------------------------------------------------------------------------
__global__ void degree_kernel(const int* __restrict__ edges,
                              float* __restrict__ deg_out,
                              float* __restrict__ deg_in) {
    int t = blockIdx.x * blockDim.x + threadIdx.x;
    if (t >= RR * EE) return;
    int r = t / EE;
    int e = t - r * EE;
    int s = edges[r * 2 * EE + e];
    int d = edges[r * 2 * EE + EE + e];
    atomicAdd(&deg_out[r * NN + s], 1.0f);
    atomicAdd(&deg_in [r * NN + d], 1.0f);
}

// In-place: v = rsqrt(max(v, 1))
__global__ void rsqrt_kernel(float* __restrict__ v, int n) {
    int i = blockIdx.x * blockDim.x + threadIdx.x;
    if (i < n) v[i] = rsqrtf(fmaxf(v[i], 1.0f));
}

// ---------------------------------------------------------------------------
// Scatter: one wave (32 lanes) per edge; each lane moves 4 floats (128 total).
// m[dst] += h[src] * rs_out[src]   via global_atomic_add_f32.
// src/dst/scale forced scalar (uniform across the wave) -> SGPR base addressing.
// m (51MB) + h (51MB) both fit in the 192MB L2 -> atomics resolve in L2.
// ---------------------------------------------------------------------------
__global__ void scatter_kernel(const int* __restrict__ src,
                               const int* __restrict__ dst,
                               const float* __restrict__ h,
                               const float* __restrict__ rs_out,
                               float* __restrict__ m) {
    int gw   = (blockIdx.x * blockDim.x + threadIdx.x) >> 5;  // wave id = edge
    int lane = threadIdx.x & 31;
    if (gw >= EE) return;
    int s = __builtin_amdgcn_readfirstlane(src[gw]);
    int d = __builtin_amdgcn_readfirstlane(dst[gw]);
    float sc = rs_out[s];   // uniform address -> scalar load
    float4 v = ((const float4*)(h + (size_t)s * DH))[lane];
    float* md = m + (size_t)d * DH + lane * 4;
    atomicAdd(md + 0, v.x * sc);
    atomicAdd(md + 1, v.y * sc);
    atomicAdd(md + 2, v.z * sc);
    atomicAdd(md + 3, v.w * sc);
}

// ---------------------------------------------------------------------------
// Convert ALL (LL*RR) 128x128 f32 weight matrices to TRANSPOSED f16 up front:
// w16t[(l*RR+r)*DH*DH + n*DH + k] = W_l_r[k*DH + n]
// ---------------------------------------------------------------------------
__global__ void convw_all_kernel(const float* __restrict__ W0,   // [R,128,128]
                                 const float* __restrict__ Wl,   // [L-1,R,128,128]
                                 _Float16* __restrict__ w16t) {
    int i = blockIdx.x * blockDim.x + threadIdx.x;
    if (i >= LL * RR * DH * DH) return;
    int mat = i >> 14;                  // (l*RR + r)
    int rem = i & (DH * DH - 1);
    int n = rem >> 7, k = rem & (DH - 1);
    int l = mat >> 2, r = mat & 3;
    const float* Wsrc = (l == 0) ? (W0 + (size_t)r * DH * DH)
                                 : (Wl + ((size_t)(l - 1) * RR + r) * DH * DH);
    w16t[i] = (_Float16)Wsrc[k * DH + n];
}

// ---------------------------------------------------------------------------
// WMMA GEMM-accumulate:  acc[16-row tile] (+)= (m * rs_in) @ W_r + b_r
// Block = 256 threads = 8 waves; block covers rows [16b, 16b+16), all 128 cols.
// Wave w owns columns [16w, 16w+16). K=128 swept in 4 steps of 32 via
// v_wmma_f32_16x16x32_f16 (f32 accumulate). W held transposed in LDS so
// both A and B fragments are contiguous 32B chunks (ds_load_b128 pairs).
// MODE is a template param -> straight-line epilogue, no dynamic branches.
// ---------------------------------------------------------------------------
template <int MODE>
__global__ void gemm_accum_kernel(const float* __restrict__ m,
                                  const float* __restrict__ rs_in,
                                  const _Float16* __restrict__ w16t,
                                  const float* __restrict__ bias,
                                  float* __restrict__ acc) {
    __shared__ _Float16 sWt[DH * DH];  // 32 KB, [n][k] (transposed)
    __shared__ _Float16 sA[16 * DH];   //  4 KB, [row][k]
    const int tid = threadIdx.x;
    const int block_row = blockIdx.x * 16;

    // Stage transposed f16 weights into LDS (16B per thread per iter)
    {
        const uint4* wsrc = (const uint4*)w16t;
        uint4*       wdst = (uint4*)sWt;
        #pragma unroll
        for (int i = tid; i < (DH * DH) / 8; i += 256) wdst[i] = wsrc[i];
    }
    // Stage A tile: scale rows by rsqrt(in-degree), convert to f16
    for (int i = tid; i < 16 * DH; i += 256) {
        int r = i >> 7, c = i & (DH - 1);
        int row = block_row + r;
        sA[i] = (_Float16)(m[(size_t)row * DH + c] * rs_in[row]);
    }
    __syncthreads();

    const int wave = tid >> 5;        // 0..7 (wave32)
    const int lane = tid & 31;
    const int mrow = lane & 15;                    // A-matrix M within tile
    const int col  = (wave << 4) | (lane & 15);    // B/D column
    const int hiA  = (lane & 16) ? 16 : 0;         // A: K-half offset (contiguous, matches B)
    const int hiB  = (lane & 16) ? 16 : 0;         // B: K-half offset (contiguous, matches A)
    const _Float16* aRow = &sA [mrow * DH + hiA];
    const _Float16* bCol = &sWt[col  * DH + hiB];

    v8f c = {};
    #pragma unroll
    for (int ks = 0; ks < DH; ks += 32) {
        // A fragment: halves 0-15 = K[ks+hiA .. +15] (contiguous)
        v8h alo = *(const v8h*)(aRow + ks);
        v8h ahi = *(const v8h*)(aRow + ks + 8);
        v16h a = __builtin_shufflevector(alo, ahi,
                     0,1,2,3,4,5,6,7,8,9,10,11,12,13,14,15);
        // B fragment: 16 contiguous K values in the transposed layout
        v16h b = *(const v16h*)(bCol + ks);
        c = __builtin_amdgcn_wmma_f32_16x16x32_f16(
            /*neg_a=*/false, a, /*neg_b=*/false, b,
            /*c_mod=*/(short)0, c, /*reuse_a=*/false, /*reuse_b=*/false);
    }

    // C/D layout: lanes 0-15 -> M = v, lanes 16-31 -> M = v+8; N = lane&15
    const int rbase = block_row + ((lane >> 4) << 3);
    const float bv = bias[col];
    float* accp = acc + (size_t)rbase * DH + col;
    #pragma unroll
    for (int v = 0; v < 8; v++) {
        float o = c[v] + bv;
        if constexpr (!(MODE & MODE_FIRST)) o += accp[(size_t)v * DH];
        if constexpr (MODE & MODE_RELU)     o = fmaxf(o, 0.0f);
        accp[(size_t)v * DH] = o;
    }
}

// ---------------------------------------------------------------------------
// Mean pooling over graphs + classifier
// ---------------------------------------------------------------------------
__global__ void pool_count_kernel(const int* __restrict__ gid, float* __restrict__ gcnt) {
    int i = blockIdx.x * blockDim.x + threadIdx.x;
    if (i < NN) atomicAdd(&gcnt[gid[i]], 1.0f);
}

__global__ void pool_sum_kernel(const float* __restrict__ h,
                                const int* __restrict__ gid,
                                float* __restrict__ gsum) {
    int t = blockIdx.x * blockDim.x + threadIdx.x;
    if (t >= NN * DH) return;
    int i = t >> 7;           // node
    int c = t & (DH - 1);     // channel
    atomicAdd(&gsum[gid[i] * DH + c], h[t]);
}

__global__ void classify_kernel(const float* __restrict__ gsum,
                                const float* __restrict__ gcnt,
                                const float* __restrict__ Wc,
                                const float* __restrict__ bc,
                                float* __restrict__ out) {
    int t = blockIdx.x * blockDim.x + threadIdx.x;
    if (t >= GG * NC) return;
    int g = t / NC, c = t - g * NC;
    float inv = 1.0f / fmaxf(gcnt[g], 1.0f);
    float a = bc[c];
    for (int k = 0; k < DH; k++) a += gsum[g * DH + k] * inv * Wc[k * NC + c];
    out[t] = a;
}

// ---------------------------------------------------------------------------
extern "C" void kernel_launch(void* const* d_in, const int* in_sizes, int n_in,
                              void* d_out, int out_size, void* d_ws, size_t ws_size,
                              hipStream_t stream) {
    const float* features = (const float*)d_in[0];   // [N,128]
    const int*   edges    = (const int*)  d_in[1];   // [R,2,E]
    const int*   gid      = (const int*)  d_in[2];   // [N]
    const float* W0       = (const float*)d_in[3];   // [R,128,128]
    const float* b0       = (const float*)d_in[4];   // [R,128]
    const float* Wl       = (const float*)d_in[5];   // [L-1,R,128,128]
    const float* bl       = (const float*)d_in[6];   // [L-1,R,128]
    const float* Wc       = (const float*)d_in[7];   // [128,10]
    const float* bc       = (const float*)d_in[8];   // [10]
    float*       out      = (float*)d_out;           // [G,10]

    // Workspace carve-up (all 256B aligned; total ~157.3 MB)
    char* ws = (char*)d_ws;
    auto take = [&](size_t bytes) {
        char* p = ws;
        ws += (bytes + 255) & ~(size_t)255;
        return p;
    };
    float*    rs_out = (float*)take((size_t)RR * NN * 4);      // rsqrt(deg_out)
    float*    rs_in  = (float*)take((size_t)RR * NN * 4);      // rsqrt(deg_in)
    float*    hA     = (float*)take((size_t)NN * DH * 4);
    float*    hB     = (float*)take((size_t)NN * DH * 4);
    float*    mbuf   = (float*)take((size_t)NN * DH * 4);
    _Float16* w16t   = (_Float16*)take((size_t)LL * RR * DH * DH * 2);
    float*    gsum   = (float*)take((size_t)GG * DH * 4 + (size_t)GG * 4);
    float*    gcnt   = gsum + GG * DH;

    const size_t featBytes = (size_t)NN * DH * 4;

    // 1) One-time prep: degrees -> rsqrt(max(deg,1)); all weights -> f16^T
    hipMemsetAsync(rs_out, 0, (size_t)2 * RR * NN * 4, stream);
    degree_kernel<<<(RR * EE + 255) / 256, 256, 0, stream>>>(edges, rs_out, rs_in);
    rsqrt_kernel<<<(2 * RR * NN + 255) / 256, 256, 0, stream>>>(rs_out, 2 * RR * NN);
    convw_all_kernel<<<(LL * RR * DH * DH + 255) / 256, 256, 0, stream>>>(W0, Wl, w16t);

    // 2) Layers
    const float* h_in = features;
    float* h_out = hA;
    for (int l = 0; l < LL; l++) {
        const float* bbase = (l == 0) ? b0 : (bl + (size_t)(l - 1) * RR * DH);
        for (int r = 0; r < RR; r++) {
            const int* src = edges + (size_t)r * 2 * EE;
            const int* dst = src + EE;
            hipMemsetAsync(mbuf, 0, featBytes, stream);
            scatter_kernel<<<(EE * 32) / 256, 256, 0, stream>>>(
                src, dst, h_in, rs_out + (size_t)r * NN, mbuf);
            const _Float16* wmat = w16t + (size_t)(l * RR + r) * DH * DH;
            const float* rsin_r  = rs_in + (size_t)r * NN;
            const float* bias_r  = bbase + (size_t)r * DH;
            if (r == 0) {
                gemm_accum_kernel<MODE_FIRST><<<NN / 16, 256, 0, stream>>>(
                    mbuf, rsin_r, wmat, bias_r, h_out);
            } else if (r == RR - 1) {
                gemm_accum_kernel<MODE_RELU><<<NN / 16, 256, 0, stream>>>(
                    mbuf, rsin_r, wmat, bias_r, h_out);
            } else {
                gemm_accum_kernel<0><<<NN / 16, 256, 0, stream>>>(
                    mbuf, rsin_r, wmat, bias_r, h_out);
            }
        }
        h_in = h_out;
        h_out = (h_out == hA) ? hB : hA;
    }
    const float* h_final = h_in;  // hA after 3 layers

    // 3) Mean pool + classifier
    hipMemsetAsync(gsum, 0, (size_t)(GG * DH + GG) * 4, stream);
    pool_count_kernel<<<(NN + 255) / 256, 256, 0, stream>>>(gid, gcnt);
    pool_sum_kernel<<<(NN * DH + 255) / 256, 256, 0, stream>>>(h_final, gid, gsum);
    classify_kernel<<<(GG * NC + 255) / 256, 256, 0, stream>>>(gsum, gcnt, Wc, bc, out);
}
